// UltraEfficientSparseFFN_67190468379264
// MI455X (gfx1250) — matrix operand order repaired and checked by reading, not compiled, not run on hardware
//
#include <hip/hip_runtime.h>
#include <math.h>

#define DD 2048
#define M_ROWS 8192
#define NBITS 11
#define KFREQ 128
#define LN_EPS 1e-5f
#define HPITCH 40  // bf16 pitch: row*20 dwords mod 64 distinct over 16 rows

typedef __attribute__((ext_vector_type(16))) __bf16 v16bf;
typedef __attribute__((ext_vector_type(16))) unsigned short v16u16;
typedef __attribute__((ext_vector_type(8))) float v8f;

union Frag { v16u16 u; v16bf b; };

static __device__ __forceinline__ unsigned short f2bf(float f) {
    unsigned u = __float_as_uint(f);
    unsigned r = u + 0x7FFFu + ((u >> 16) & 1u);
    return (unsigned short)(r >> 16);
}
static __device__ __forceinline__ float bf2f(unsigned short h) {
    return __uint_as_float(((unsigned)h) << 16);
}

// ---- CDNA5 async global->LDS (ASYNCcnt-tracked DMA, no VGPR data) ----
static __device__ __forceinline__ unsigned lds_addr_of(const void* p) {
    return (unsigned)(unsigned long long)p;  // low 32 bits = LDS byte address
}
static __device__ __forceinline__ void async_load_b128(unsigned lds, const void* g) {
    asm volatile("global_load_async_to_lds_b128 %0, %1, off"
                 :
                 : "v"(lds), "v"((unsigned long long)g)
                 : "memory");
}
static __device__ __forceinline__ void wait_async0() {
    asm volatile("s_wait_asynccnt 0x0" ::: "memory");
}

static __device__ __forceinline__ float blk_reduce(float v, float* red, int tid) {
    red[tid] = v;
    __syncthreads();
#pragma unroll
    for (int s = 128; s > 0; s >>= 1) {
        if (tid < s) red[tid] += red[tid + s];
        __syncthreads();
    }
    float r = red[0];
    __syncthreads();
    return r;
}

// ---------------------------------------------------------------------------
// Kernel 0: static top-k masks over poly_imp (keep 1024) and micro_imp (512).
// ---------------------------------------------------------------------------
__global__ __launch_bounds__(256) void mask_kernel(const float* __restrict__ poly_imp,
                                                   const float* __restrict__ micro_imp,
                                                   float* __restrict__ pmask,
                                                   float* __restrict__ mmask) {
    __shared__ unsigned s_u[DD];
    __shared__ int s_cnt;
    int tid = threadIdx.x;
    for (int pass = 0; pass < 2; ++pass) {
        const float* imp = pass ? micro_imp : poly_imp;
        float* mask = pass ? mmask : pmask;
        int K = pass ? 512 : 1024;
        for (int i = tid; i < DD; i += 256) {
            unsigned u = __float_as_uint(imp[i]);
            u = (u & 0x80000000u) ? ~u : (u | 0x80000000u);
            s_u[i] = u;
        }
        __syncthreads();
        unsigned prefix = 0;
        int want = K;
        for (int bit = 31; bit >= 0; --bit) {
            if (tid == 0) s_cnt = 0;
            __syncthreads();
            unsigned hi_mask = (bit == 31) ? 0u : ~((1u << (bit + 1)) - 1u);
            int local = 0;
            for (int i = tid; i < DD; i += 256) {
                unsigned u = s_u[i];
                if ((u & hi_mask) == prefix && ((u >> bit) & 1u)) local++;
            }
            if (local) atomicAdd(&s_cnt, local);
            __syncthreads();
            int c = s_cnt;
            __syncthreads();
            if (c >= want) prefix |= (1u << bit);
            else want -= c;
        }
        for (int i = tid; i < DD; i += 256) {
            unsigned u = s_u[i];
            float m = 0.f;
            if (u > prefix) m = 1.f;
            else if (u == prefix) {
                int order = 0;
                for (int j = 0; j < i; j++)
                    if (s_u[j] == prefix) order++;
                if (order < want) m = 1.f;
            }
            mask[i] = m;
        }
        __syncthreads();
    }
}

// ---------------------------------------------------------------------------
// Kernel 0b: one-shot split of W_res into bf16 hi/lo planes (row-major [e][d]).
// ---------------------------------------------------------------------------
__global__ __launch_bounds__(256) void wsplit_kernel(const float* __restrict__ W,
                                                     unsigned short* __restrict__ whi,
                                                     unsigned short* __restrict__ wlo) {
    size_t i = ((size_t)blockIdx.x * 256 + threadIdx.x) * 4;
    float4 q = *(const float4*)(W + i);
    float v[4] = {q.x, q.y, q.z, q.w};
    unsigned short h[4], l[4];
#pragma unroll
    for (int e = 0; e < 4; e++) {
        h[e] = f2bf(v[e]);
        l[e] = f2bf(v[e] - bf2f(h[e]));
    }
    unsigned* ph = (unsigned*)(whi + i);
    unsigned* pl = (unsigned*)(wlo + i);
    ph[0] = (unsigned)h[0] | ((unsigned)h[1] << 16);
    ph[1] = (unsigned)h[2] | ((unsigned)h[3] << 16);
    pl[0] = (unsigned)l[0] | ((unsigned)l[1] << 16);
    pl[1] = (unsigned)l[2] | ((unsigned)l[3] << 16);
}

// ---------------------------------------------------------------------------
// In-place radix-2 DIT FFT of 2048 complex points in LDS.
// ---------------------------------------------------------------------------
static __device__ void fft2048(float* re, float* im, const float* twr, const float* twi,
                               int tid) {
    for (int i = tid; i < DD; i += 256) {
        int j = (int)(__brev((unsigned)i) >> (32 - NBITS));
        if (i < j) {
            float tr = re[i]; re[i] = re[j]; re[j] = tr;
            float ti = im[i]; im[i] = im[j]; im[j] = ti;
        }
    }
    __syncthreads();
    for (int st = 1; st <= NBITS; ++st) {
        int half = 1 << (st - 1);
        int tstep = DD >> st;
        for (int b = tid; b < DD / 2; b += 256) {
            int pos = b & (half - 1);
            int grp = b >> (st - 1);
            int i0 = (grp << st) + pos;
            int i1 = i0 + half;
            float c = twr[pos * tstep];
            float s = twi[pos * tstep];
            float xr = re[i1], xi = im[i1];
            float tr = xr * c - xi * s;
            float ti = xr * s + xi * c;
            float ar = re[i0], ai = im[i0];
            re[i0] = ar + tr; im[i0] = ai + ti;
            re[i1] = ar - tr; im[i1] = ai - ti;
        }
        __syncthreads();
    }
}

// ---------------------------------------------------------------------------
// Kernel 1: one block per row. LN-in -> FFT -> top-128 spectral gate -> iFFT
// -> poly/micro on masked lanes -> LN-out -> bf16 hi/lo planes (ws).
// ---------------------------------------------------------------------------
__global__ __launch_bounds__(256) void row_kernel(
    const float* __restrict__ x, const float* __restrict__ ln_in_g,
    const float* __restrict__ ln_in_b, const float* __restrict__ ln_out_g,
    const float* __restrict__ ln_out_b, const float* __restrict__ spec_gains,
    const float* __restrict__ spec_bias, const float* __restrict__ poly_coeffs,
    const float* __restrict__ w1p, const float* __restrict__ b1p,
    const float* __restrict__ w2p, const float* __restrict__ b2p,
    const float* __restrict__ pmask, const float* __restrict__ mmask,
    unsigned short* __restrict__ hhi, unsigned short* __restrict__ hlo) {
    __shared__ float re[DD];
    __shared__ float im[DD];
    __shared__ float twr[DD / 2];
    __shared__ float twi[DD / 2];
    __shared__ float mag[DD / 2 + 1];
    __shared__ float scl[DD / 2 + 1];
    __shared__ float red[256];
    __shared__ unsigned long long keys[256];
    __shared__ int s_cnt;

    const int tid = threadIdx.x;
    const int row = blockIdx.x;
    const float* xr = x + (size_t)row * DD;

    // DMA the row into LDS (re[]) while building the twiddle table.
#pragma unroll
    for (int i = 0; i < 2; i++) {
        int idx = tid + 256 * i;
        async_load_b128(lds_addr_of(&re[idx * 4]), xr + idx * 4);
    }
    for (int k = tid; k < DD / 2; k += 256) {
        float s, c;
        __sincosf(-6.28318530717958647692f * (float)k * (1.0f / (float)DD), &s, &c);
        twr[k] = c;
        twi[k] = s;
    }
    keys[tid] = 0ull;
    wait_async0();
    __syncthreads();

    // ---- LayerNorm in ----
    float v[8];
    float msum = 0.f;
#pragma unroll
    for (int i = 0; i < 8; i++) {
        float t = re[tid + 256 * i];
        v[i] = t;
        msum += t;
    }
    float mu = blk_reduce(msum, red, tid) * (1.f / DD);
    float msq = 0.f;
#pragma unroll
    for (int i = 0; i < 8; i++) {
        float d = v[i] - mu;
        msq += d * d;
    }
    float rstd = rsqrtf(blk_reduce(msq, red, tid) * (1.f / DD) + LN_EPS);
#pragma unroll
    for (int i = 0; i < 8; i++) {
        int d = tid + 256 * i;
        re[d] = (v[i] - mu) * rstd * ln_in_g[d] + ln_in_b[d];
        im[d] = 0.f;
    }
    __syncthreads();

    fft2048(re, im, twr, twi, tid);

    for (int k = tid; k <= DD / 2; k += 256) {
        mag[k] = sqrtf(re[k] * re[k] + im[k] * im[k]);
        scl[k] = 0.f;
    }
    __syncthreads();

    // ---- radix-select 128th largest magnitude ----
    unsigned prefix = 0;
    int want = KFREQ;
    for (int bit = 30; bit >= 0; --bit) {
        if (tid == 0) s_cnt = 0;
        __syncthreads();
        unsigned hi_mask = ~((1u << (bit + 1)) - 1u);
        int local = 0;
        for (int k = tid; k <= DD / 2; k += 256) {
            unsigned u = __float_as_uint(mag[k]);
            if ((u & hi_mask) == prefix && ((u >> bit) & 1u)) local++;
        }
        if (local) atomicAdd(&s_cnt, local);
        __syncthreads();
        int c = s_cnt;
        __syncthreads();
        if (c >= want) prefix |= (1u << bit);
        else want -= c;
    }

    // ---- gather candidates, bitonic sort desc by (mag, -idx) ----
    if (tid == 0) s_cnt = 0;
    __syncthreads();
    for (int k = tid; k <= DD / 2; k += 256) {
        unsigned u = __float_as_uint(mag[k]);
        if (u >= prefix) {
            int p = atomicAdd(&s_cnt, 1);
            if (p < 256)
                keys[p] = ((unsigned long long)u << 32) | (unsigned)(0xFFFFFFFFu - (unsigned)k);
        }
    }
    __syncthreads();
    for (int k2 = 2; k2 <= 256; k2 <<= 1) {
        for (int j = k2 >> 1; j > 0; j >>= 1) {
            int ixj = tid ^ j;
            if (ixj > tid) {
                unsigned long long a = keys[tid];
                unsigned long long b = keys[ixj];
                bool descRegion = ((tid & k2) == 0);
                bool sw = descRegion ? (a < b) : (a > b);
                if (sw) { keys[tid] = b; keys[ixj] = a; }
            }
            __syncthreads();
        }
    }
    if (tid < KFREQ) {
        unsigned long long key = keys[tid];
        if (key != 0ull) {
            int idx = (int)(0xFFFFFFFFu - (unsigned)(key & 0xFFFFFFFFu));
            scl[idx] = spec_gains[tid];
        }
    }
    __syncthreads();

    // ---- spectral gate + Hermitian mirror ----
    for (int k = tid; k <= DD / 2; k += 256) {
        float s = scl[k];
        re[k] *= s;
        im[k] *= s;
    }
    __syncthreads();
    for (int k = tid + 1; k < DD / 2; k += 256) {
        re[DD - k] = re[k];
        im[DD - k] = -im[k];
    }
    __syncthreads();

    // ---- inverse FFT via conj trick ----
    for (int i = tid; i < DD; i += 256) im[i] = -im[i];
    __syncthreads();
    fft2048(re, im, twr, twi, tid);

    // ---- pointwise + LN-out, emit bf16 hi/lo planes for the WMMA GEMM ----
    const float c0 = poly_coeffs[0], c1 = poly_coeffs[1], c2 = poly_coeffs[2];
    const float w1 = *w1p, b1 = *b1p, w2 = *w2p, b2 = *b2p;
    float hv[8];
    float hsum = 0.f;
#pragma unroll
    for (int i = 0; i < 8; i++) {
        int d = tid + 256 * i;
        float h = re[d] * (1.f / (float)DD) + spec_bias[d];
        float acc = ((c2 * h + c1) * h + c0) * h;
        h = (pmask[d] > 0.f) ? acc : h;
        float y = w1 * h + b1;
        y = y / (1.f + __expf(-y));
        y = w2 * y + b2;
        y = y / (1.f + __expf(-y));
        h = (mmask[d] > 0.f) ? y : h;
        hv[i] = h;
        hsum += h;
    }
    float mu2 = blk_reduce(hsum, red, tid) * (1.f / DD);
    float sq2 = 0.f;
#pragma unroll
    for (int i = 0; i < 8; i++) {
        float d = hv[i] - mu2;
        sq2 += d * d;
    }
    float rstd2 = rsqrtf(blk_reduce(sq2, red, tid) * (1.f / DD) + LN_EPS);
    unsigned short* oh = hhi + (size_t)row * DD;
    unsigned short* ol = hlo + (size_t)row * DD;
#pragma unroll
    for (int i = 0; i < 8; i++) {
        int d = tid + 256 * i;
        float o = (hv[i] - mu2) * rstd2 * ln_out_g[d] + ln_out_b[d];
        unsigned short h = f2bf(o);
        oh[d] = h;
        ol[d] = f2bf(o - bf2f(h));
    }
}

// ---------------------------------------------------------------------------
// Kernel 2: out = x + gate * (H @ W^T + b_res), bf16x3 WMMA.
// Async DMA of pre-split bf16 tiles, double-buffered; inner loop is pure
// ds_load_b128 + v_wmma (no conversion VALU).
// ---------------------------------------------------------------------------
static __device__ __forceinline__ v8f wmma_bf16(const Frag& a, const Frag& b, v8f c) {
    return __builtin_amdgcn_wmma_f32_16x16x32_bf16(false, a.b, false, b.b, (short)0, c,
                                                   false, false);
}

// one 128x32 bf16 tile = 512 16B chunks; each thread issues 2 async loads
static __device__ __forceinline__ void stage_tile16(const unsigned short* base, int kk,
                                                    unsigned short (*buf)[HPITCH],
                                                    int tid) {
#pragma unroll
    for (int i = 0; i < 2; i++) {
        int idx = tid + 256 * i;
        int row = idx >> 2;   // 4 chunks of 8 halfs per 32-half row
        int chunk = idx & 3;
        const unsigned short* g = base + (size_t)row * DD + kk + chunk * 8;
        async_load_b128(lds_addr_of(&buf[row][chunk * 8]), g);
    }
}

__global__ __launch_bounds__(256) void gemm_kernel(
    const unsigned short* __restrict__ Hhi, const unsigned short* __restrict__ Hlo,
    const unsigned short* __restrict__ Whi, const unsigned short* __restrict__ Wlo,
    const float* __restrict__ x, const float* __restrict__ b_res,
    const float* __restrict__ gatep, float* __restrict__ out) {
    __shared__ unsigned short Ah[2][128][HPITCH];  // 2 x 10KB each array
    __shared__ unsigned short Al[2][128][HPITCH];
    __shared__ unsigned short Bh[2][128][HPITCH];
    __shared__ unsigned short Bl[2][128][HPITCH];

    const int tid = threadIdx.x;
    const int lane = tid & 31;
    const int wave = tid >> 5;
    const int wm = wave >> 2;  // 0..1 -> 64 rows
    const int wn = wave & 3;   // 0..3 -> 32 cols
    const int m0 = blockIdx.y * 128;
    const int n0 = blockIdx.x * 128;
    const int lm = lane & 15;
    const int koffA = (lane < 16) ? 0 : 8;
    const int koffB = (lane < 16) ? 0 : 16;

    const unsigned short* baseAh = Hhi + (size_t)m0 * DD;
    const unsigned short* baseAl = Hlo + (size_t)m0 * DD;
    const unsigned short* baseBh = Whi + (size_t)n0 * DD;
    const unsigned short* baseBl = Wlo + (size_t)n0 * DD;

    const v8f vzero = {0.f, 0.f, 0.f, 0.f, 0.f, 0.f, 0.f, 0.f};
    v8f acc[4][2];
#pragma unroll
    for (int mi = 0; mi < 4; mi++)
#pragma unroll
        for (int ni = 0; ni < 2; ni++) acc[mi][ni] = vzero;

    // prologue: DMA first K-tile
    stage_tile16(baseAh, 0, Ah[0], tid);
    stage_tile16(baseAl, 0, Al[0], tid);
    stage_tile16(baseBh, 0, Bh[0], tid);
    stage_tile16(baseBl, 0, Bl[0], tid);

    int buf = 0;
    for (int kk = 0; kk < DD; kk += 32) {
        wait_async0();
        __syncthreads();  // tile visible; previous buffer fully consumed

        if (kk + 32 < DD) {
            stage_tile16(baseAh, kk + 32, Ah[buf ^ 1], tid);
            stage_tile16(baseAl, kk + 32, Al[buf ^ 1], tid);
            stage_tile16(baseBh, kk + 32, Bh[buf ^ 1], tid);
            stage_tile16(baseBl, kk + 32, Bl[buf ^ 1], tid);
        }

        // B fragments: lane = column; lanes<16 K=0..15, lanes>=16 K=16..31
        Frag bh[2], bl[2];
#pragma unroll
        for (int ni = 0; ni < 2; ni++) {
            int bRow = wn * 32 + ni * 16 + lm;
            const unsigned short* ph = &Bh[buf][bRow][koffB];
            const unsigned short* pl = &Bl[buf][bRow][koffB];
#pragma unroll
            for (int j = 0; j < 16; j++) {
                bh[ni].u[j] = ph[j];
                bl[ni].u[j] = pl[j];
            }
        }
        // A fragments: lane = row; lanes<16 K=0..15, lanes>=16 K=16..31
#pragma unroll
        for (int mi = 0; mi < 4; mi++) {
            int aRow = wm * 64 + mi * 16 + lm;
            const unsigned short* ph = &Ah[buf][aRow][0];
            const unsigned short* pl = &Al[buf][aRow][0];
            Frag ah, al;
#pragma unroll
            for (int j = 0; j < 8; j++) {
                ah.u[j] = ph[koffB + j];
                ah.u[8 + j] = ph[koffB + 8 + j];
                al.u[j] = pl[koffB + j];
                al.u[8 + j] = pl[koffB + 8 + j];
            }
#pragma unroll
            for (int ni = 0; ni < 2; ni++) {
                acc[mi][ni] = wmma_bf16(ah, bh[ni], acc[mi][ni]);  // hi*hi
                acc[mi][ni] = wmma_bf16(ah, bl[ni], acc[mi][ni]);  // hi*lo
                acc[mi][ni] = wmma_bf16(al, bh[ni], acc[mi][ni]);  // lo*hi
            }
        }
        buf ^= 1;
    }

    // epilogue: out = x + gate * (acc + b_res)
    const float gate = *gatep;
#pragma unroll
    for (int mi = 0; mi < 4; mi++) {
#pragma unroll
        for (int ni = 0; ni < 2; ni++) {
            int nl = n0 + wn * 32 + ni * 16 + lm;
            float br = b_res[nl];
#pragma unroll
            for (int e = 0; e < 8; e++) {
                int ml = m0 + wm * 64 + mi * 16 + e + ((lane >= 16) ? 8 : 0);
                size_t off = (size_t)ml * DD + nl;
                out[off] = x[off] + gate * (acc[mi][ni][e] + br);
            }
        }
    }
}

// ---------------------------------------------------------------------------
extern "C" void kernel_launch(void* const* d_in, const int* in_sizes, int n_in,
                              void* d_out, int out_size, void* d_ws, size_t ws_size,
                              hipStream_t stream) {
    const float* x          = (const float*)d_in[0];
    const float* ln_in_g    = (const float*)d_in[1];
    const float* ln_in_b    = (const float*)d_in[2];
    const float* ln_out_g   = (const float*)d_in[3];
    const float* ln_out_b   = (const float*)d_in[4];
    const float* spec_gains = (const float*)d_in[5];
    const float* spec_bias  = (const float*)d_in[6];
    const float* poly_coef  = (const float*)d_in[7];
    const float* poly_imp   = (const float*)d_in[8];
    const float* micro_imp  = (const float*)d_in[9];
    const float* micro_w1   = (const float*)d_in[10];
    const float* micro_b1   = (const float*)d_in[11];
    const float* micro_w2   = (const float*)d_in[12];
    const float* micro_b2   = (const float*)d_in[13];
    const float* W_res      = (const float*)d_in[14];
    const float* b_res      = (const float*)d_in[15];
    const float* gate       = (const float*)d_in[16];

    // workspace layout (16B-aligned blocks)
    char* wsb = (char*)d_ws;
    float* pmask = (float*)wsb;                               // 2048 f32
    float* mmask = (float*)(wsb + 8192);                      // 2048 f32
    unsigned short* whi = (unsigned short*)(wsb + 16384);     // 4M bf16
    unsigned short* wlo = whi + (size_t)DD * DD;              // 4M bf16
    unsigned short* hhi = wlo + (size_t)DD * DD;              // 16M bf16
    unsigned short* hlo = hhi + (size_t)M_ROWS * DD;          // 16M bf16

    mask_kernel<<<1, 256, 0, stream>>>(poly_imp, micro_imp, pmask, mmask);

    wsplit_kernel<<<(DD * DD) / (256 * 4), 256, 0, stream>>>(W_res, whi, wlo);

    row_kernel<<<M_ROWS, 256, 0, stream>>>(x, ln_in_g, ln_in_b, ln_out_g, ln_out_b,
                                           spec_gains, spec_bias, poly_coef, micro_w1,
                                           micro_b1, micro_w2, micro_b2, pmask, mmask,
                                           hhi, hlo);

    dim3 grid(DD / 128, M_ROWS / 128);
    gemm_kernel<<<grid, 256, 0, stream>>>(hhi, hlo, whi, wlo, x, b_res, gate,
                                          (float*)d_out);
}
